// ConvFCNet_62423054680334
// MI455X (gfx1250) — matrix operand order repaired and checked by reading, not compiled, not run on hardware
//
#include <hip/hip_runtime.h>

typedef __attribute__((ext_vector_type(16))) _Float16 v16h;
typedef __attribute__((ext_vector_type(8)))  _Float16 v8h;
typedef __attribute__((ext_vector_type(8)))  float    v8f;

#define BATCH 512

// ---------------------------------------------------------------------------
// WMMA fragment helpers (layouts per CDNA5 ISA 7.12.2, wave32)
// A frag 16x32: lanes<16 hold K{0-7,16-23}, lanes>=16 hold K{8-15,24-31}
//   -> caller passes p = rowbase + hiK*8 ; halves = p[0..7], (p+16)[0..7]
// B frag 32x16: same K split as A (hiK*8 offset)
//   -> caller passes p = colbase + hiK*8 ; halves = p[0..7], (p+16)[0..7]
// ---------------------------------------------------------------------------
__device__ __forceinline__ v16h load_frag_a(const _Float16* __restrict__ p) {
    v8h lo = *(const v8h*)(p);
    v8h hi = *(const v8h*)(p + 16);
    v16h a;
#pragma unroll
    for (int j = 0; j < 8; ++j) { a[j] = lo[j]; a[8 + j] = hi[j]; }
    return a;
}
__device__ __forceinline__ v16h load_frag_b(const _Float16* __restrict__ p) {
    v8h lo = *(const v8h*)(p);
    v8h hi = *(const v8h*)(p + 16);
    v16h b;
#pragma unroll
    for (int j = 0; j < 8; ++j) { b[j] = lo[j]; b[8 + j] = hi[j]; }
    return b;
}
__device__ __forceinline__ v16h zero_frag() {
    v16h z;
#pragma unroll
    for (int j = 0; j < 16; ++j) z[j] = (_Float16)0.f;
    return z;
}

// ---------------------------------------------------------------------------
// fake-quant helpers
// ---------------------------------------------------------------------------
__global__ void zero_slots_kernel(unsigned* slots) {
    if (threadIdx.x < 3) slots[threadIdx.x] = 0u;
}

__global__ __launch_bounds__(256)
void maxabs_kernel(const float* __restrict__ w, int n, unsigned* __restrict__ slot) {
    __shared__ unsigned red[256];
    int t = threadIdx.x;
    float m = 0.f;
    for (int i = blockIdx.x * blockDim.x + t; i < n; i += gridDim.x * blockDim.x)
        m = fmaxf(m, fabsf(w[i]));
    red[t] = __float_as_uint(m);   // non-negative floats: uint order == float order
    __syncthreads();
    for (int s = 128; s > 0; s >>= 1) {
        if (t < s) red[t] = red[t] > red[t + s] ? red[t] : red[t + s];
        __syncthreads();
    }
    if (t == 0) atomicMax(slot, red[0]);
}

__device__ __forceinline__ float fq(float w, float maxabs) {
    float scale = maxabs * (1.f / 127.f);
    if (!(scale > 0.f)) return w;
    float q = rintf(w / scale);              // round-to-nearest-even == jnp.round
    q = fminf(fmaxf(q, -127.f), 127.f);
    return q * scale;
}

// W1 [512,4608] -> quantize -> f16, permuted so K index matches NHWC feature flatten
__global__ __launch_bounds__(256)
void quant_w1_kernel(const float* __restrict__ W1, const unsigned* __restrict__ slot,
                     _Float16* __restrict__ wT1) {
    int idx = blockIdx.x * blockDim.x + threadIdx.x;          // 512*4608
    if (idx >= 512 * 4608) return;
    float ma = __uint_as_float(*slot);
    int n = idx / 4608, k = idx % 4608;                       // k in NCHW order c*36+h*6+w
    int cc = k / 36, rem = k % 36, hh = rem / 6, ww = rem % 6;
    int k_nhwc = (hh * 6 + ww) * 128 + cc;
    wT1[(size_t)n * 4608 + k_nhwc] = (_Float16)fq(W1[idx], ma);
}

__global__ __launch_bounds__(256)
void quant_f32_kernel(const float* __restrict__ W, const unsigned* __restrict__ slot,
                      float* __restrict__ out, int n) {
    int idx = blockIdx.x * blockDim.x + threadIdx.x;
    if (idx >= n) return;
    out[idx] = fq(W[idx], __uint_as_float(*slot));
}

// conv weights OIHW f32 -> [9][COUT][CIN] f16  (B-matrix friendly: row n, contiguous K)
__global__ __launch_bounds__(256)
void prep_convw_kernel(const float* __restrict__ w, _Float16* __restrict__ wT,
                       int COUT, int CIN) {
    int idx = blockIdx.x * blockDim.x + threadIdx.x;          // (tap*COUT+co)*CIN+ci
    int total = 9 * COUT * CIN;
    if (idx >= total) return;
    int ci  = idx % CIN;
    int co  = (idx / CIN) % COUT;
    int tap = idx / (CIN * COUT);
    int r = tap / 3, s = tap % 3;
    wT[idx] = (_Float16)w[((co * CIN + ci) * 3 + r) * 3 + s];
}

// ---------------------------------------------------------------------------
// conv1: direct f32, Cin=3 (too small for WMMA K=32), weights staged in LDS.
// x NCHW [512,3,48,48] -> out NHWC f16 [512,48,48,32] with bias+ReLU fused.
// ---------------------------------------------------------------------------
__global__ __launch_bounds__(256)
void conv1_direct_kernel(const float* __restrict__ x, const float* __restrict__ w,
                         const float* __restrict__ bias, _Float16* __restrict__ out) {
    __shared__ float sw[32 * 27 + 32];
    int t = threadIdx.x;
    for (int i = t; i < 32 * 27; i += 256) sw[i] = w[i];      // OIHW contiguous = [co][ci*9+r*3+s]
    if (t < 32) sw[32 * 27 + t] = bias[t];
    __syncthreads();

    int idx = blockIdx.x * blockDim.x + t;                    // pixel id, 512*48*48 (divisible)
    int ow = idx % 48, oh = (idx / 48) % 48, b = idx / (48 * 48);

    float acc[32];
#pragma unroll
    for (int co = 0; co < 32; ++co) acc[co] = sw[32 * 27 + co];

    for (int ci = 0; ci < 3; ++ci) {
#pragma unroll
        for (int r = 0; r < 3; ++r) {
            int ih = oh + r - 1;
            if ((unsigned)ih >= 48u) continue;
#pragma unroll
            for (int s = 0; s < 3; ++s) {
                int iw = ow + s - 1;
                if ((unsigned)iw >= 48u) continue;
                float xv = x[((size_t)(b * 3 + ci) * 48 + ih) * 48 + iw];
                int wo = ci * 9 + r * 3 + s;
#pragma unroll
                for (int co = 0; co < 32; ++co) acc[co] += xv * sw[co * 27 + wo];
            }
        }
    }
    _Float16* op = out + (size_t)idx * 32;
#pragma unroll
    for (int co = 0; co < 32; ++co) {
        float v = acc[co];
        op[co] = (_Float16)(v > 0.f ? v : 0.f);
    }
}

// ---------------------------------------------------------------------------
// 2x2 maxpool stride 2, NHWC f16
// ---------------------------------------------------------------------------
__global__ __launch_bounds__(256)
void maxpool_kernel(const _Float16* __restrict__ in, _Float16* __restrict__ out,
                    int H, int C, int total) {
    int idx = blockIdx.x * blockDim.x + threadIdx.x;
    if (idx >= total) return;
    int Wo = H >> 1;
    int c  = idx % C;
    int r  = idx / C;
    int ow = r % Wo, oh = (r / Wo) % Wo, b = r / (Wo * Wo);
    size_t base = ((size_t)(b * H + oh * 2) * H + ow * 2) * C + c;
    size_t rs = (size_t)H * C;
    float m0 = (float)in[base],      m1 = (float)in[base + C];
    float m2 = (float)in[base + rs], m3 = (float)in[base + rs + C];
    float m = fmaxf(fmaxf(m0, m1), fmaxf(m2, m3));
    out[idx] = (_Float16)m;
}

// ---------------------------------------------------------------------------
// 3x3 conv as 9 shifted WMMA GEMMs. NHWC activations, [9][COUT][CIN] weights.
// 32(M=pixels) x 32(N=channels) register tile per wave: 2 A-frags x 2 B-frags
// -> 4 v_wmma per load group (2:1 load:wmma), f32 accum, bias+ReLU fused.
// ---------------------------------------------------------------------------
template<int CIN, int COUT, int HW>
__global__ __launch_bounds__(256)
void conv3x3_wmma_kernel(const _Float16* __restrict__ in,   // [B][HW][HW][CIN]
                         const _Float16* __restrict__ wT,   // [9][COUT][CIN]
                         const float*    __restrict__ bias, // [COUT]
                         _Float16*       __restrict__ out)  // [B][HW][HW][COUT]
{
    constexpr int M  = BATCH * HW * HW;
    constexpr int NT = COUT / 32;
    constexpr int MT = M / 32;
    const int wave = blockIdx.x * 8 + (threadIdx.x >> 5);
    if (wave >= MT * NT) return;                            // wave-uniform exit
    const int lane = threadIdx.x & 31;
    const int tn = wave % NT, tm = wave / NT;
    const int row = lane & 15;                              // A row for loads / D column
    const int hiK = lane >> 4;
    const int n0  = tn * 32 + row;
    const int n1  = n0 + 16;

    const int mA0 = tm * 32 + row;                          // A-frag0 row
    const int mA1 = mA0 + 16;                               // A-frag1 row
    const int ow0 = mA0 % HW, oh0 = (mA0 / HW) % HW, b0 = mA0 / (HW * HW);
    const int ow1 = mA1 % HW, oh1 = (mA1 / HW) % HW, b1 = mA1 / (HW * HW);

    v8f c00, c01, c10, c11;
    const float bv0 = bias[n0], bv1 = bias[n1];
#pragma unroll
    for (int i = 0; i < 8; ++i) { c00[i] = bv0; c01[i] = bv1; c10[i] = bv0; c11[i] = bv1; }

#pragma unroll
    for (int tap = 0; tap < 9; ++tap) {
        const int dr = tap / 3 - 1, ds = tap % 3 - 1;
        const int ih0 = oh0 + dr, iw0 = ow0 + ds;
        const int ih1 = oh1 + dr, iw1 = ow1 + ds;
        const bool inb0 = ((unsigned)ih0 < (unsigned)HW) && ((unsigned)iw0 < (unsigned)HW);
        const bool inb1 = ((unsigned)ih1 < (unsigned)HW) && ((unsigned)iw1 < (unsigned)HW);
        const _Float16* wtap = wT + (size_t)tap * COUT * CIN;
#pragma unroll
        for (int kc = 0; kc < CIN / 32; ++kc) {
            v16h a0 = inb0
                ? load_frag_a(in + ((size_t)(b0 * HW + ih0) * HW + iw0) * CIN + kc * 32 + hiK * 8)
                : zero_frag();
            v16h a1 = inb1
                ? load_frag_a(in + ((size_t)(b1 * HW + ih1) * HW + iw1) * CIN + kc * 32 + hiK * 8)
                : zero_frag();
            v16h bb0 = load_frag_b(wtap + (size_t)n0 * CIN + kc * 32 + hiK * 8);
            v16h bb1 = load_frag_b(wtap + (size_t)n1 * CIN + kc * 32 + hiK * 8);
            c00 = __builtin_amdgcn_wmma_f32_16x16x32_f16(false, a0, false, bb0, (short)0, c00, false, false);
            c01 = __builtin_amdgcn_wmma_f32_16x16x32_f16(false, a0, false, bb1, (short)0, c01, false, false);
            c10 = __builtin_amdgcn_wmma_f32_16x16x32_f16(false, a1, false, bb0, (short)0, c10, false, false);
            c11 = __builtin_amdgcn_wmma_f32_16x16x32_f16(false, a1, false, bb1, (short)0, c11, false, false);
        }
    }
    // D layout: VGPR i -> row i (lanes<16) / row i+8 (lanes>=16), column = lane&15
#pragma unroll
    for (int i = 0; i < 8; ++i) {
        {   // row group 0 (c00 col n0, c01 col n1) share one pixel decode
            int mo = tm * 32 + i + hiK * 8;
            int ow = mo % HW, oh = (mo / HW) % HW, b = mo / (HW * HW);
            _Float16* op = out + ((size_t)(b * HW + oh) * HW + ow) * COUT;
            float v0 = c00[i]; v0 = v0 > 0.f ? v0 : 0.f;
            float v1 = c01[i]; v1 = v1 > 0.f ? v1 : 0.f;
            op[n0] = (_Float16)v0;
            op[n1] = (_Float16)v1;
        }
        {   // row group 1 (c10, c11)
            int mo = tm * 32 + 16 + i + hiK * 8;
            int ow = mo % HW, oh = (mo / HW) % HW, b = mo / (HW * HW);
            _Float16* op = out + ((size_t)(b * HW + oh) * HW + ow) * COUT;
            float v0 = c10[i]; v0 = v0 > 0.f ? v0 : 0.f;
            float v1 = c11[i]; v1 = v1 > 0.f ? v1 : 0.f;
            op[n0] = (_Float16)v0;
            op[n1] = (_Float16)v1;
        }
    }
}

// ---------------------------------------------------------------------------
// FC1: cur1[512,512] = feat[512,4608] @ W1q^T   (WMMA, 32x32 wave tile)
// ---------------------------------------------------------------------------
__global__ __launch_bounds__(256)
void fc1_wmma_kernel(const _Float16* __restrict__ feat,  // [512][4608]
                     const _Float16* __restrict__ wT1,   // [512][4608]
                     float* __restrict__ cur1)           // [512][512]
{
    const int wave = blockIdx.x * 8 + (threadIdx.x >> 5);
    if (wave >= 16 * 16) return;
    const int lane = threadIdx.x & 31;
    const int tn = wave & 15, tm = wave >> 4;
    const int row = lane & 15;
    const int hiK = lane >> 4;
    const int n0  = tn * 32 + row;
    const int m0  = tm * 32 + row;

    v8f c00, c01, c10, c11;
#pragma unroll
    for (int i = 0; i < 8; ++i) { c00[i] = 0.f; c01[i] = 0.f; c10[i] = 0.f; c11[i] = 0.f; }

    const _Float16* ap0 = feat + (size_t)m0 * 4608;
    const _Float16* ap1 = ap0 + (size_t)16 * 4608;
    const _Float16* bp0 = wT1 + (size_t)n0 * 4608;
    const _Float16* bp1 = bp0 + (size_t)16 * 4608;
    for (int k0 = 0; k0 < 4608; k0 += 32) {
        __builtin_prefetch(ap0 + k0 + 256, 0, 0);   // global_prefetch_b8
        __builtin_prefetch(bp0 + k0 + 256, 0, 0);
        v16h a0  = load_frag_a(ap0 + k0 + hiK * 8);
        v16h a1  = load_frag_a(ap1 + k0 + hiK * 8);
        v16h bb0 = load_frag_b(bp0 + k0 + hiK * 8);
        v16h bb1 = load_frag_b(bp1 + k0 + hiK * 8);
        c00 = __builtin_amdgcn_wmma_f32_16x16x32_f16(false, a0, false, bb0, (short)0, c00, false, false);
        c01 = __builtin_amdgcn_wmma_f32_16x16x32_f16(false, a0, false, bb1, (short)0, c01, false, false);
        c10 = __builtin_amdgcn_wmma_f32_16x16x32_f16(false, a1, false, bb0, (short)0, c10, false, false);
        c11 = __builtin_amdgcn_wmma_f32_16x16x32_f16(false, a1, false, bb1, (short)0, c11, false, false);
    }
#pragma unroll
    for (int i = 0; i < 8; ++i) {
        int moA = tm * 32 + i + hiK * 8;
        int moB = moA + 16;
        cur1[(size_t)moA * 512 + n0]      = c00[i];
        cur1[(size_t)moA * 512 + n0 + 16] = c01[i];
        cur1[(size_t)moB * 512 + n0]      = c10[i];
        cur1[(size_t)moB * 512 + n0 + 16] = c11[i];
    }
}

// ---------------------------------------------------------------------------
// LIF head: 3 timesteps. One block per batch row; spikes staged in LDS.
// ---------------------------------------------------------------------------
__global__ __launch_bounds__(128)
void lif_head_kernel(const float* __restrict__ cur1, const float* __restrict__ W2q,
                     const float* __restrict__ W3q, float* __restrict__ out) {
    __shared__ float s1[512];
    __shared__ float s2[128];
    __shared__ float v3acc[10];   // [0..4]=v3 state, [5..9]=spike accumulator
    const int b = blockIdx.x, t = threadIdx.x;

    float v1[4] = {0.f, 0.f, 0.f, 0.f};
    float v2 = 0.f;
    if (t < 10) v3acc[t] = 0.f;
    __syncthreads();

    for (int step = 0; step < 3; ++step) {
        // layer 1: 512 neurons, 4 per thread
#pragma unroll
        for (int i = 0; i < 4; ++i) {
            int k = t * 4 + i;
            float v = 0.5f * (v1[i] + cur1[(size_t)b * 512 + k]);   // v + (cur-v)/tau, tau=2
            float sp = (v >= 1.f) ? 1.f : 0.f;
            v1[i] = v * (1.f - sp);
            s1[k] = sp;
        }
        __syncthreads();
        // layer 2: thread t = neuron t
        float cur2 = 0.f;
        for (int k = 0; k < 512; ++k) cur2 += s1[k] * W2q[(size_t)t * 512 + k];
        float v = 0.5f * (v2 + cur2);
        float sp = (v >= 1.f) ? 1.f : 0.f;
        v2 = v * (1.f - sp);
        s2[t] = sp;
        __syncthreads();
        // layer 3: 5 neurons
        if (t < 5) {
            float cur3 = 0.f;
            for (int j = 0; j < 128; ++j) cur3 += s2[j] * W3q[t * 128 + j];
            float v3 = 0.5f * (v3acc[t] + cur3);
            float sp3 = (v3 >= 1.f) ? 1.f : 0.f;
            v3acc[t] = v3 * (1.f - sp3);
            v3acc[5 + t] += sp3;
        }
        __syncthreads();
    }
    if (t < 5) out[b * 5 + t] = v3acc[5 + t] * (1.f / 3.f);
}

// ---------------------------------------------------------------------------
// host side
// ---------------------------------------------------------------------------
extern "C" void kernel_launch(void* const* d_in, const int* in_sizes, int n_in,
                              void* d_out, int out_size, void* d_ws, size_t ws_size,
                              hipStream_t stream) {
    (void)in_sizes; (void)n_in; (void)out_size; (void)ws_size;
    const float* x       = (const float*)d_in[0];
    const float* conv1_w = (const float*)d_in[1];
    const float* conv1_b = (const float*)d_in[2];
    const float* conv2_w = (const float*)d_in[3];
    const float* conv2_b = (const float*)d_in[4];
    const float* conv3_w = (const float*)d_in[5];
    const float* conv3_b = (const float*)d_in[6];
    const float* W1      = (const float*)d_in[7];
    const float* W2      = (const float*)d_in[8];
    const float* W3      = (const float*)d_in[9];
    float* out = (float*)d_out;

    // workspace carve-up (256B aligned)
    unsigned char* ws = (unsigned char*)d_ws;
    size_t off = 0;
    auto carve = [&](size_t bytes) -> void* {
        void* p = ws + off;
        off = (off + bytes + 255) & ~(size_t)255;
        return p;
    };
    unsigned*  umax  = (unsigned*) carve(3 * sizeof(unsigned));
    _Float16*  wT1   = (_Float16*) carve((size_t)512 * 4608 * 2);
    float*     W2q   = (float*)    carve((size_t)128 * 512 * 4);
    float*     W3q   = (float*)    carve((size_t)5 * 128 * 4);
    _Float16*  wT2   = (_Float16*) carve((size_t)9 * 64 * 32 * 2);
    _Float16*  wT3   = (_Float16*) carve((size_t)9 * 128 * 64 * 2);
    _Float16*  c1out = (_Float16*) carve((size_t)BATCH * 48 * 48 * 32 * 2);
    _Float16*  feat1 = (_Float16*) carve((size_t)BATCH * 24 * 24 * 32 * 2);
    _Float16*  c2out = (_Float16*) carve((size_t)BATCH * 24 * 24 * 64 * 2);
    _Float16*  feat2 = (_Float16*) carve((size_t)BATCH * 12 * 12 * 64 * 2);
    _Float16*  c3out = (_Float16*) carve((size_t)BATCH * 12 * 12 * 128 * 2);
    _Float16*  feat3 = (_Float16*) carve((size_t)BATCH * 6 * 6 * 128 * 2);
    float*     cur1  = (float*)    carve((size_t)BATCH * 512 * 4);

    // --- fake-quant FC weights ---
    zero_slots_kernel<<<1, 32, 0, stream>>>(umax);
    {
        int n1 = 512 * 4608, n2 = 128 * 512, n3 = 5 * 128;
        int g1 = (n1 + 255) / 256; if (g1 > 2048) g1 = 2048;
        maxabs_kernel<<<g1, 256, 0, stream>>>(W1, n1, umax + 0);
        maxabs_kernel<<<(n2 + 255) / 256, 256, 0, stream>>>(W2, n2, umax + 1);
        maxabs_kernel<<<(n3 + 255) / 256, 256, 0, stream>>>(W3, n3, umax + 2);
        quant_w1_kernel<<<(n1 + 255) / 256, 256, 0, stream>>>(W1, umax + 0, wT1);
        quant_f32_kernel<<<(n2 + 255) / 256, 256, 0, stream>>>(W2, umax + 1, W2q, n2);
        quant_f32_kernel<<<(n3 + 255) / 256, 256, 0, stream>>>(W3, umax + 2, W3q, n3);
    }
    // --- conv weight relayout (f16, [9][COUT][CIN]) ---
    prep_convw_kernel<<<(9 * 64 * 32 + 255) / 256, 256, 0, stream>>>(conv2_w, wT2, 64, 32);
    prep_convw_kernel<<<(9 * 128 * 64 + 255) / 256, 256, 0, stream>>>(conv3_w, wT3, 128, 64);

    // --- conv1 (direct f32) + pool ---
    conv1_direct_kernel<<<(BATCH * 48 * 48) / 256, 256, 0, stream>>>(x, conv1_w, conv1_b, c1out);
    {
        int tot = BATCH * 24 * 24 * 32;
        maxpool_kernel<<<(tot + 255) / 256, 256, 0, stream>>>(c1out, feat1, 48, 32, tot);
    }
    // --- conv2 (WMMA, 32x32 wave tiles) + pool ---
    {
        int waves = (BATCH * 24 * 24 / 32) * (64 / 32);       // 18432
        conv3x3_wmma_kernel<32, 64, 24><<<waves / 8, 256, 0, stream>>>(feat1, wT2, conv2_b, c2out);
        int tot = BATCH * 12 * 12 * 64;
        maxpool_kernel<<<(tot + 255) / 256, 256, 0, stream>>>(c2out, feat2, 24, 64, tot);
    }
    // --- conv3 (WMMA, 32x32 wave tiles) + pool ---
    {
        int waves = (BATCH * 12 * 12 / 32) * (128 / 32);      // 9216
        conv3x3_wmma_kernel<64, 128, 12><<<waves / 8, 256, 0, stream>>>(feat2, wT3, conv3_b, c3out);
        int tot = BATCH * 6 * 6 * 128;
        maxpool_kernel<<<(tot + 255) / 256, 256, 0, stream>>>(c3out, feat3, 12, 128, tot);
    }
    // --- FC1 (WMMA, 32x32 wave tiles) ---
    fc1_wmma_kernel<<<(16 * 16) / 8, 256, 0, stream>>>(feat3, wT1, cur1);
    // --- LIF head, 3 timesteps ---
    lif_head_kernel<<<BATCH, 128, 0, stream>>>(cur1, W2q, W3q, out);
}
